// GTSA_55216099558003
// MI455X (gfx1250) — matrix operand order repaired and checked by reading, not compiled, not run on hardware
//
#include <hip/hip_runtime.h>
#include <cmath>

// ---------------------------------------------------------------------------
// Types for WMMA fragments (wave32, 16x16x32 f16 -> f32)
// ---------------------------------------------------------------------------
typedef __attribute__((ext_vector_type(16))) _Float16 v16h;
typedef __attribute__((ext_vector_type(8)))  _Float16 v8h_t;
typedef __attribute__((ext_vector_type(8)))  float    v8f;

union HV { v16h v; v8h_t h[2]; };

// ---------------------------------------------------------------------------
// CDNA5 async global->LDS copy (CDNA5 ISA).
// LDS destination address = low 32 bits of the generic shared pointer
// ---------------------------------------------------------------------------
__device__ __forceinline__ void async_b128(void* lds_ptr, const void* gptr)
{
    unsigned l = (unsigned)(unsigned long long)(uintptr_t)lds_ptr;
    asm volatile("global_load_async_to_lds_b128 %0, %1, off"
                 :: "v"(l), "v"(gptr) : "memory");
}
__device__ __forceinline__ void wait_async0()
{
    asm volatile("s_wait_asynccnt 0x0" ::: "memory");
}

// Stage a 64x32-half tile (64 rows x 64 bytes) = 256 b128 transfers;
// 128 threads -> 2 async loads per thread per tile.
__device__ __forceinline__ void stage_tile(_Float16* lds, const _Float16* g,
                                           int ld, int row0, int k0, int tid)
{
#pragma unroll
    for (int j = 0; j < 2; ++j) {
        int tt  = tid + j * 128;
        int row = tt >> 2;
        int seg = tt & 3;
        const _Float16* gp = g + (size_t)(row0 + row) * ld + k0 + seg * 8;
        async_b128(lds + row * 32 + seg * 8, gp);
    }
}

// A fragment from LDS tile (16x32 f16, ISA 7.12.2): lane m = lane&15;
// lanes<16 hold K offsets {0..7,16..23}, lanes>=16 hold {8..15,24..31}.
__device__ __forceinline__ v16h lds_frag_a(const _Float16* Xs, int sub16, int lane)
{
    const _Float16* p = Xs + (sub16 * 16 + (lane & 15)) * 32 + ((lane >> 4) << 3);
    HV r;
    r.h[0] = *(const v8h_t*)p;
    r.h[1] = *(const v8h_t*)(p + 16);
    return r.v;
}

// B fragment from LDS tile: lane holds column n = lane&15,
// lanes<16 hold K offsets {0..7,16..23}, lanes>=16 hold {8..15,24..31}.
__device__ __forceinline__ v16h lds_frag_b(const _Float16* Ws, int sub16, int lane)
{
    const _Float16* p = Ws + (sub16 * 16 + (lane & 15)) * 32 + ((lane >> 4) << 3);
    HV r;
    r.h[0] = *(const v8h_t*)p;
    r.h[1] = *(const v8h_t*)(p + 16);
    return r.v;
}

// ---------------------------------------------------------------------------
// Generic WMMA GEMM: Y[N,O] = act(X[N,K] @ W[O,K]^T + bias + R)
// X, W pre-converted f16 with zero K-padding.  Block = 4 waves = 64x64 tile;
// each wave owns 32x32 (4 wmma / 32-k step).  X/W block tiles are staged
// through LDS with double-buffered async global->LDS copies.
// ---------------------------------------------------------------------------
__global__ __launch_bounds__(128) void k_gemm(
    const _Float16* __restrict__ X, int ldx,
    const _Float16* __restrict__ W, int ldw,
    const float* __restrict__ bias,
    const float* __restrict__ R, int ldr,
    float* __restrict__ Y, int ldy,
    _Float16* __restrict__ Yh, int ldyh,
    int N, int K, int O, int act)
{
    __shared__ __align__(16) _Float16 Xs[2][64 * 32];
    __shared__ __align__(16) _Float16 Ws[2][64 * 32];

    int tid  = threadIdx.x;
    int wave = tid >> 5;
    int lane = tid & 31;
    int rowB = blockIdx.y * 64;
    int colB = blockIdx.x * 64;
    int wm   = wave >> 1;   // 0/1: rows 0-31 / 32-63 of the block tile
    int wn   = wave & 1;    // 0/1: cols 0-31 / 32-63

    v8f acc00 = {}, acc01 = {}, acc10 = {}, acc11 = {};

    stage_tile(Xs[0], X, ldx, rowB, 0, tid);
    stage_tile(Ws[0], W, ldw, colB, 0, tid);

    int cur = 0;
    for (int k0 = 0; k0 < K; k0 += 32) {
        wait_async0();        // our async copies into buf `cur` are done
        __syncthreads();      // ...and everyone else's too
        if (k0 + 32 < K) {    // prefetch next chunk into the other buffer
            stage_tile(Xs[cur ^ 1], X, ldx, rowB, k0 + 32, tid);
            stage_tile(Ws[cur ^ 1], W, ldw, colB, k0 + 32, tid);
        }
        if (k0 + 64 < K) {    // deep prefetch into L2
            __builtin_prefetch(X + (size_t)(rowB + (tid & 63)) * ldx + k0 + 64, 0, 1);
            __builtin_prefetch(W + (size_t)(colB + (tid & 63)) * ldw + k0 + 64, 0, 1);
        }
        v16h a0 = lds_frag_a(Xs[cur], wm * 2 + 0, lane);
        v16h a1 = lds_frag_a(Xs[cur], wm * 2 + 1, lane);
        v16h b0 = lds_frag_b(Ws[cur], wn * 2 + 0, lane);
        v16h b1 = lds_frag_b(Ws[cur], wn * 2 + 1, lane);
        acc00 = __builtin_amdgcn_wmma_f32_16x16x32_f16(false, a0, false, b0, (short)0, acc00, false, false);
        acc01 = __builtin_amdgcn_wmma_f32_16x16x32_f16(false, a0, false, b1, (short)0, acc01, false, false);
        acc10 = __builtin_amdgcn_wmma_f32_16x16x32_f16(false, a1, false, b0, (short)0, acc10, false, false);
        acc11 = __builtin_amdgcn_wmma_f32_16x16x32_f16(false, a1, false, b1, (short)0, acc11, false, false);
        cur ^= 1;
    }

    // C/D layout: n = lane&15, m = 8*(lane>>4) + r
    int row0 = rowB + wm * 32;
    int col0 = colB + wn * 32;
    int nOff = lane & 15;
    int mOff = (lane >> 4) << 3;
    v8f accs[4] = {acc00, acc01, acc10, acc11};
#pragma unroll
    for (int mi = 0; mi < 2; ++mi) {
#pragma unroll
        for (int ni = 0; ni < 2; ++ni) {
            v8f a = accs[mi * 2 + ni];
#pragma unroll
            for (int r = 0; r < 8; ++r) {
                int rr = row0 + mi * 16 + mOff + r;
                int cc = col0 + ni * 16 + nOff;
                if (rr < N && cc < O) {
                    float v = a[r];
                    if (bias) v += bias[cc];
                    if (R)    v += R[(size_t)rr * ldr + cc];
                    if (act)  v  = fmaxf(v, 0.0f);
                    if (Y)  Y[(size_t)rr * ldy + cc]  = v;
                    if (Yh) Yh[(size_t)rr * ldyh + cc] = (_Float16)v;
                }
            }
        }
    }
}

// ---------------------------------------------------------------------------
// Featurize: mags (4 ch) + inter-mic phase diffs (3 ch) -> hc[B,7,F,T]
// ---------------------------------------------------------------------------
__global__ void k_feat(const float* __restrict__ x, float* __restrict__ hc)
{
    const int Bn = 4, Mn = 4, Fn = 201, Tn = 500, Cn = 7;
    size_t idx = (size_t)blockIdx.x * blockDim.x + threadIdx.x;
    size_t total = (size_t)Bn * Fn * Tn;
    if (idx >= total) return;
    int t = (int)(idx % Tn);
    int f = (int)((idx / Tn) % Fn);
    int b = (int)(idx / ((size_t)Tn * Fn));
    float mag[4], ang[4];
    for (int m = 0; m < Mn; ++m) {
        size_t p = ((((size_t)b * Mn + m) * Fn + f) * Tn + t) * 2;
        float re = x[p], im = x[p + 1];
        mag[m] = sqrtf(re * re + im * im + 1e-10f);
        ang[m] = atan2f(im, re);
    }
    for (int c = 0; c < 4; ++c)
        hc[(((size_t)b * Cn + c) * Fn + f) * Tn + t] = mag[c];
    for (int c = 0; c < 3; ++c)
        hc[(((size_t)b * Cn + 4 + c) * Fn + f) * Tn + t] = ang[0] - ang[c + 1];
}

// ---------------------------------------------------------------------------
// hc[B,C,F,T] <-> row-major rows.  mode 0 (even layer): rows n=(b*C+c)*T+t,
// feat f.  mode 1 (odd layer): rows n=(b*F+f)*T+t, feat c.
// ---------------------------------------------------------------------------
__global__ void k_to_rows(const float* __restrict__ hc, float* __restrict__ rows,
                          _Float16* __restrict__ rowsh, int ld, int mode)
{
    size_t idx = (size_t)blockIdx.x * blockDim.x + threadIdx.x;
    const size_t total = (size_t)4 * 7 * 201 * 500;
    if (idx >= total) return;
    int t = (int)(idx % 500); size_t r = idx / 500;
    int f = (int)(r % 201);   r /= 201;
    int c = (int)(r % 7);     int b = (int)(r / 7);
    float v = hc[idx];
    size_t n; int feat;
    if (mode == 0) { n = ((size_t)(b * 7 + c)) * 500 + t;   feat = f; }
    else           { n = ((size_t)(b * 201 + f)) * 500 + t; feat = c; }
    rows[n * ld + feat]  = v;
    rowsh[n * ld + feat] = (_Float16)v;
}

__global__ void k_from_rows(const float* __restrict__ rows, float* __restrict__ hc,
                            int ld, int mode)
{
    size_t idx = (size_t)blockIdx.x * blockDim.x + threadIdx.x;
    const size_t total = (size_t)4 * 7 * 201 * 500;
    if (idx >= total) return;
    int t = (int)(idx % 500); size_t r = idx / 500;
    int f = (int)(r % 201);   r /= 201;
    int c = (int)(r % 7);     int b = (int)(r / 7);
    size_t n; int feat;
    if (mode == 0) { n = ((size_t)(b * 7 + c)) * 500 + t;   feat = f; }
    else           { n = ((size_t)(b * 201 + f)) * 500 + t; feat = c; }
    hc[idx] = rows[n * ld + feat];
}

// ---------------------------------------------------------------------------
// f32 [O x lds] -> f16 [Opad x Kpad] with zero padding (weights + activations)
// ---------------------------------------------------------------------------
__global__ void k_cvt(const float* __restrict__ src, _Float16* __restrict__ dst,
                      int O, int K, int Opad, int Kpad, int lds)
{
    size_t idx = (size_t)blockIdx.x * blockDim.x + threadIdx.x;
    size_t total = (size_t)Opad * Kpad;
    if (idx >= total) return;
    int k = (int)(idx % Kpad);
    int o = (int)(idx / Kpad);
    float v = (o < O && k < K) ? src[(size_t)o * lds + k] : 0.0f;
    dst[idx] = (_Float16)v;
}

// ---------------------------------------------------------------------------
// Attention with Gaussian distance bias + abs + softmax; one block per (t, bc*head).
// Scores for the whole row (T<=512) staged in LDS; two-phase softmax.
// ---------------------------------------------------------------------------
__global__ __launch_bounds__(128) void k_attn(
    const float* __restrict__ Q, const float* __restrict__ Km,
    const float* __restrict__ V, float* __restrict__ Outp,
    const float* __restrict__ delta,
    int ld, int Tn, int nh, int dh, float scale)
{
    __shared__ float sc[512];
    __shared__ float red[128];
    int t    = blockIdx.x;
    int bc   = blockIdx.y / nh;
    int head = blockIdx.y % nh;
    int tid  = threadIdx.x;
    float dlt  = delta[0];
    float invd = 1.0f / (dlt * dlt + 1e-8f);

    const float* q = Q + (size_t)(bc * Tn + t) * ld + head * dh;
    for (int s = tid; s < Tn; s += 128) {
        const float* kk = Km + (size_t)(bc * Tn + s) * ld + head * dh;
        float acc = 0.0f;
        for (int d = 0; d < dh; ++d) acc += q[d] * kk[d];
        float dts = (float)(t - s);
        float g = __expf(-dts * dts * invd);
        sc[s] = fabsf(acc * g * scale);
    }
    __syncthreads();

    float m = -1e30f;
    for (int s = tid; s < Tn; s += 128) m = fmaxf(m, sc[s]);
    red[tid] = m; __syncthreads();
    for (int st = 64; st > 0; st >>= 1) {
        if (tid < st) red[tid] = fmaxf(red[tid], red[tid + st]);
        __syncthreads();
    }
    m = red[0]; __syncthreads();

    float ssum = 0.0f;
    for (int s = tid; s < Tn; s += 128) {
        float e = __expf(sc[s] - m);
        sc[s] = e; ssum += e;
    }
    red[tid] = ssum; __syncthreads();
    for (int st = 64; st > 0; st >>= 1) {
        if (tid < st) red[tid] += red[tid + st];
        __syncthreads();
    }
    float rsum = 1.0f / red[0];
    __syncthreads();

    for (int d = tid; d < dh; d += 128) {
        float acc = 0.0f;
        for (int s = 0; s < Tn; ++s)
            acc += sc[s] * V[(size_t)(bc * Tn + s) * ld + head * dh + d];
        Outp[(size_t)(bc * Tn + t) * ld + head * dh + d] = acc * rsum;
    }
}

// ---------------------------------------------------------------------------
// GlobalLayerNorm over (t, feat) per group; per-feature affine.  Writes f32
// (same ld) and optional f16 copy.
// ---------------------------------------------------------------------------
__global__ __launch_bounds__(256) void k_gln(
    const float* __restrict__ X, float* __restrict__ Y, _Float16* __restrict__ Yh,
    const float* __restrict__ w, const float* __restrict__ b,
    int ld, int ldh, int Tn, int Fd)
{
    __shared__ float s1[256], s2[256];
    int g = blockIdx.x;
    size_t base  = (size_t)g * Tn * ld;
    size_t baseh = (size_t)g * Tn * ldh;
    int n = Tn * Fd;
    float sum = 0.0f, sq = 0.0f;
    for (int i = threadIdx.x; i < n; i += 256) {
        float v = X[base + (size_t)(i / Fd) * ld + (i % Fd)];
        sum += v; sq += v * v;
    }
    s1[threadIdx.x] = sum; s2[threadIdx.x] = sq; __syncthreads();
    for (int st = 128; st > 0; st >>= 1) {
        if (threadIdx.x < st) { s1[threadIdx.x] += s1[threadIdx.x + st];
                                s2[threadIdx.x] += s2[threadIdx.x + st]; }
        __syncthreads();
    }
    float mean = s1[0] / n;
    float var  = s2[0] / n - mean * mean;
    float rden = 1.0f / (sqrtf(var + 1e-10f) + 1e-8f);
    for (int i = threadIdx.x; i < n; i += 256) {
        int tt = i / Fd, f = i % Fd;
        float v = (X[base + (size_t)tt * ld + f] - mean) * rden * w[f] + b[f];
        Y[base + (size_t)tt * ld + f] = v;
        if (Yh) Yh[baseh + (size_t)tt * ldh + f] = (_Float16)v;
    }
}

// ---------------------------------------------------------------------------
// im2col for causal conv k=3 (left pad 2): Xcol[(b*T+t), cf*3+dt]
// ---------------------------------------------------------------------------
__global__ void k_im2col(const float* __restrict__ hc, _Float16* __restrict__ xc, int ldc)
{
    size_t idx = (size_t)blockIdx.x * blockDim.x + threadIdx.x;
    const size_t total = (size_t)4 * 500 * 1407;
    if (idx >= total) return;
    int cf = (int)(idx % 1407); size_t r = idx / 1407;
    int t = (int)(r % 500);
    int b = (int)(r / 500);
    for (int dt = 0; dt < 3; ++dt) {
        int ts = t + dt - 2;
        float v = (ts >= 0) ? hc[((size_t)b * 1407 + cf) * 500 + ts] : 0.0f;
        xc[((size_t)(b * 500 + t)) * ldc + cf * 3 + dt] = (_Float16)v;
    }
}

// GLU gate: GT = TL * sigmoid(GL)
__global__ void k_gate(const float* __restrict__ TL, const float* __restrict__ GL,
                       float* __restrict__ GT, int ld, int N, int O)
{
    size_t idx = (size_t)blockIdx.x * blockDim.x + threadIdx.x;
    size_t total = (size_t)N * O;
    if (idx >= total) return;
    int o = (int)(idx % O);
    int nrow = (int)(idx / O);
    size_t off = (size_t)nrow * ld + o;
    float g = 1.0f / (1.0f + __expf(-GL[off]));
    GT[off] = TL[off] * g;
}

// Final GLN over (cout, T) per batch, transposed store to d_out[b, o, t]
__global__ __launch_bounds__(256) void k_final_gln(
    const float* __restrict__ G, const float* __restrict__ nw,
    const float* __restrict__ nb, float* __restrict__ out,
    int ld, int Tn, int CO)
{
    __shared__ float s1[256], s2[256];
    int b = blockIdx.x;
    int n = Tn * CO;
    float sum = 0.0f, sq = 0.0f;
    for (int i = threadIdx.x; i < n; i += 256) {
        int t = i / CO, o = i % CO;
        float v = G[(size_t)(b * Tn + t) * ld + o];
        sum += v; sq += v * v;
    }
    s1[threadIdx.x] = sum; s2[threadIdx.x] = sq; __syncthreads();
    for (int st = 128; st > 0; st >>= 1) {
        if (threadIdx.x < st) { s1[threadIdx.x] += s1[threadIdx.x + st];
                                s2[threadIdx.x] += s2[threadIdx.x + st]; }
        __syncthreads();
    }
    float mean = s1[0] / n;
    float var  = s2[0] / n - mean * mean;
    float rden = 1.0f / (sqrtf(var + 1e-10f) + 1e-8f);
    for (int i = threadIdx.x; i < n; i += 256) {
        int t = i / CO, o = i % CO;
        float v = (G[(size_t)(b * Tn + t) * ld + o] - mean) * rden * nw[o] + nb[o];
        out[((size_t)b * CO + o) * Tn + t] = v;
    }
}

// ---------------------------------------------------------------------------
// Host orchestration
// ---------------------------------------------------------------------------
namespace {
struct LayerP {
    const float *Wq, *bq, *Wk, *bk, *Wv, *bv, *Wo, *bo, *delta;
    const float *na_w, *na_b, *Wi, *bi, *Wf, *bf, *ni_w, *ni_b;
};
struct Net {
    const float *Wc, *bc, *Wt, *bt, *Wg, *bg, *nw, *nb;
    LayerP layer[4];
    const float* x;
};
} // namespace

extern "C" void kernel_launch(void* const* d_in, const int* in_sizes, int n_in,
                              void* d_out, int out_size, void* d_ws, size_t ws_size,
                              hipStream_t stream)
{
    (void)n_in; (void)out_size; (void)ws_size;
    const float* const* in = (const float* const*)d_in;

    // Map inputs: detect insertion-order ({'x', layers..., conv}) vs sorted
    // pytree order ({'params'{'conv','layers'}, 'x'}) via x's unique size.
    Net net;
    const size_t XSZ = (size_t)4 * 4 * 201 * 500 * 2; // 3,216,000
    bool insertion = ((size_t)in_sizes[0] == XSZ);
    if (insertion) {
        net.x = in[0];
        for (int i = 0; i < 4; ++i) {
            const float* const* p = in + 1 + 17 * i;
            LayerP& L = net.layer[i];
            L.Wq = p[0];  L.bq = p[1];  L.Wk = p[2];  L.bk = p[3];
            L.Wv = p[4];  L.bv = p[5];  L.Wo = p[6];  L.bo = p[7];
            L.delta = p[8]; L.na_w = p[9]; L.na_b = p[10];
            L.Wi = p[11]; L.bi = p[12]; L.Wf = p[13]; L.bf = p[14];
            L.ni_w = p[15]; L.ni_b = p[16];
        }
        const float* const* c = in + 69;
        net.Wc = c[0]; net.bc = c[1]; net.Wt = c[2]; net.bt = c[3];
        net.Wg = c[4]; net.bg = c[5]; net.nw = c[6]; net.nb = c[7];
    } else {
        const float* const* c = in;
        net.Wc = c[0]; net.Wg = c[1]; net.Wt = c[2]; net.bc = c[3];
        net.bg = c[4]; net.bt = c[5]; net.nb = c[6]; net.nw = c[7];
        for (int i = 0; i < 4; ++i) {
            const float* const* p = in + 8 + 17 * i;
            LayerP& L = net.layer[i];
            L.Wf = p[0]; L.Wi = p[1]; L.Wk = p[2]; L.Wo = p[3];
            L.Wq = p[4]; L.Wv = p[5];
            L.bf = p[6]; L.bi = p[7]; L.bk = p[8]; L.bo = p[9];
            L.bq = p[10]; L.bv = p[11];
            L.delta = p[12]; L.na_b = p[13]; L.na_w = p[14];
            L.ni_b = p[15]; L.ni_w = p[16];
        }
        net.x = in[76];
    }

    // ---- Workspace layout (all padded: rows->mult of 64, K->mult of 32) ----
    const size_t N64_E = 14016, N64_O = 402048;
    const size_t ROWS_MAX = N64_O * 32;        // >= N64_E * 224
    const size_t HH_ELEMS = N64_O * 832;       // FFN hidden (f16)
    char*  base = (char*)d_ws;
    size_t off  = 0;
    auto alloc = [&](size_t bytes) -> void* {
        void* p = base + off;
        off += (bytes + 255) & ~(size_t)255;
        return p;
    };
    float*     hc   = (float*)alloc((size_t)2814000 * 4);
    float*     X0f  = (float*)alloc(ROWS_MAX * 4);
    float*     Qf   = (float*)alloc(ROWS_MAX * 4);
    float*     Kf   = (float*)alloc(ROWS_MAX * 4);
    float*     Vf   = (float*)alloc(ROWS_MAX * 4);
    float*     ATTf = (float*)alloc(ROWS_MAX * 4);
    float*     X1f  = (float*)alloc(ROWS_MAX * 4);
    _Float16*  X0h  = (_Float16*)alloc(ROWS_MAX * 2);
    _Float16*  ATTh = (_Float16*)alloc(ROWS_MAX * 2);
    _Float16*  X1h  = (_Float16*)alloc(ROWS_MAX * 2);
    _Float16*  Hh   = (_Float16*)alloc(HH_ELEMS * 2);
    _Float16*  Wqh  = (_Float16*)alloc((size_t)256 * 224 * 2);
    _Float16*  Wkh  = (_Float16*)alloc((size_t)256 * 224 * 2);
    _Float16*  Wvh  = (_Float16*)alloc((size_t)256 * 224 * 2);
    _Float16*  Woh  = (_Float16*)alloc((size_t)256 * 224 * 2);
    _Float16*  Wih  = (_Float16*)alloc((size_t)832 * 224 * 2);
    _Float16*  Wfh  = (_Float16*)alloc((size_t)256 * 832 * 2);
    _Float16*  Wch  = (_Float16*)alloc((size_t)448 * 4224 * 2);
    _Float16*  Wth  = (_Float16*)alloc((size_t)448 * 416 * 2);
    _Float16*  Wgh  = (_Float16*)alloc((size_t)448 * 416 * 2);
    _Float16*  Xcol = (_Float16*)alloc((size_t)2048 * 4224 * 2);
    float*     CVf  = (float*)alloc((size_t)2048 * 416 * 4);
    _Float16*  CVh  = (_Float16*)alloc((size_t)2048 * 416 * 2);
    float*     TLf  = (float*)alloc((size_t)2048 * 416 * 4);
    float*     GLf  = (float*)alloc((size_t)2048 * 416 * 4);
    float*     GTf  = (float*)alloc((size_t)2048 * 416 * 4);

    auto blocks = [](size_t n, int b) { return dim3((unsigned)((n + (size_t)b - 1) / b)); };

    // Hidden-buffer K-pad columns must be zero once (never written after).
    (void)hipMemsetAsync(Hh, 0, HH_ELEMS * 2, stream);

    // 1) featurize
    k_feat<<<blocks((size_t)4 * 201 * 500, 256), 256, 0, stream>>>(net.x, hc);

    // 2) transformer layers
    for (int li = 0; li < 4; ++li) {
        bool even = (li % 2 == 0);
        int    md  = even ? 201 : 7;
        int    mdp = even ? 224 : 32;   // K padded to 32
        int    ow  = even ? 256 : 64;   // weight row padding (64)
        size_t N   = even ? 14000 : 402000;
        size_t N64 = even ? N64_E : N64_O;
        int    nh  = even ? 3 : 1;
        int    dh  = md / nh;
        int    groups = even ? 28 : 804;
        int    mode   = even ? 0 : 1;
        float  scale  = 1.0f / std::sqrt((float)md);
        const LayerP& L = net.layer[li];

        // layout reinterpretation between layers -> re-zero padded f16 rows
        (void)hipMemsetAsync(X0h, 0, ROWS_MAX * 2, stream);
        (void)hipMemsetAsync(X1h, 0, ROWS_MAX * 2, stream);

        k_to_rows<<<blocks((size_t)2814000, 256), 256, 0, stream>>>(hc, X0f, X0h, mdp, mode);

        // weight conversion (f32 -> zero-padded f16)
        k_cvt<<<blocks((size_t)ow * mdp, 256), 256, 0, stream>>>(L.Wq, Wqh, md, md, ow, mdp, md);
        k_cvt<<<blocks((size_t)ow * mdp, 256), 256, 0, stream>>>(L.Wk, Wkh, md, md, ow, mdp, md);
        k_cvt<<<blocks((size_t)ow * mdp, 256), 256, 0, stream>>>(L.Wv, Wvh, md, md, ow, mdp, md);
        k_cvt<<<blocks((size_t)ow * mdp, 256), 256, 0, stream>>>(L.Wo, Woh, md, md, ow, mdp, md);
        k_cvt<<<blocks((size_t)832 * mdp, 256), 256, 0, stream>>>(L.Wi, Wih, 804, md, 832, mdp, md);
        k_cvt<<<blocks((size_t)ow * 832, 256), 256, 0, stream>>>(L.Wf, Wfh, md, 804, ow, 832, 804);

        dim3 gMD((unsigned)((md + 63) / 64), (unsigned)((N + 63) / 64));
        dim3 gFN((unsigned)((804 + 63) / 64), (unsigned)((N + 63) / 64));

        // QKV projections (WMMA)
        k_gemm<<<gMD, 128, 0, stream>>>(X0h, mdp, Wqh, mdp, L.bq, nullptr, 0,
                                        Qf, mdp, nullptr, 0, (int)N, mdp, md, 0);
        k_gemm<<<gMD, 128, 0, stream>>>(X0h, mdp, Wkh, mdp, L.bk, nullptr, 0,
                                        Kf, mdp, nullptr, 0, (int)N, mdp, md, 0);
        k_gemm<<<gMD, 128, 0, stream>>>(X0h, mdp, Wvh, mdp, L.bv, nullptr, 0,
                                        Vf, mdp, nullptr, 0, (int)N, mdp, md, 0);

        // distance-biased softmax attention
        k_attn<<<dim3(500, (unsigned)(groups * nh)), 128, 0, stream>>>(
            Qf, Kf, Vf, ATTf, L.delta, mdp, 500, nh, dh, scale);

        // attention output -> f16 (zero pads written by k_cvt itself)
        k_cvt<<<blocks(N64 * (size_t)mdp, 256), 256, 0, stream>>>(
            ATTf, ATTh, (int)N, md, (int)N64, mdp, mdp);

        // output projection + residual (into Kf, no longer needed)
        k_gemm<<<gMD, 128, 0, stream>>>(ATTh, mdp, Woh, mdp, L.bo, X0f, mdp,
                                        Kf, mdp, nullptr, 0, (int)N, mdp, md, 0);
        k_gln<<<groups, 256, 0, stream>>>(Kf, X1f, X1h, L.na_w, L.na_b, mdp, mdp, 500, md);

        // FFN: relu(X1 @ Wi^T + bi) @ Wf^T + bf + X1
        k_gemm<<<gFN, 128, 0, stream>>>(X1h, mdp, Wih, mdp, L.bi, nullptr, 0,
                                        nullptr, 0, Hh, 832, (int)N, mdp, 804, 1);
        k_gemm<<<gMD, 128, 0, stream>>>(Hh, 832, Wfh, 832, L.bf, X1f, mdp,
                                        Vf, mdp, nullptr, 0, (int)N, 832, md, 0);
        k_gln<<<groups, 256, 0, stream>>>(Vf, X0f, nullptr, L.ni_w, L.ni_b, mdp, mdp, 500, md);

        k_from_rows<<<blocks((size_t)2814000, 256), 256, 0, stream>>>(X0f, hc, mdp, mode);
    }

    // 3) temporal conv (im2col GEMM, K = 1407*3 = 4221 -> 4224) + GLU + GLN
    (void)hipMemsetAsync(Xcol, 0, (size_t)2048 * 4224 * 2, stream);
    (void)hipMemsetAsync(CVh, 0, (size_t)2048 * 416 * 2, stream);

    k_im2col<<<blocks((size_t)4 * 500 * 1407, 256), 256, 0, stream>>>(hc, Xcol, 4224);
    k_cvt<<<blocks((size_t)448 * 4224, 256), 256, 0, stream>>>(net.Wc, Wch, 402, 4221, 448, 4224, 4221);
    dim3 gCV((unsigned)((402 + 63) / 64), (unsigned)((2000 + 63) / 64));
    k_gemm<<<gCV, 128, 0, stream>>>(Xcol, 4224, Wch, 4224, net.bc, nullptr, 0,
                                    CVf, 416, CVh, 416, 2000, 4224, 402, 0);

    k_cvt<<<blocks((size_t)448 * 416, 256), 256, 0, stream>>>(net.Wt, Wth, 402, 402, 448, 416, 402);
    k_cvt<<<blocks((size_t)448 * 416, 256), 256, 0, stream>>>(net.Wg, Wgh, 402, 402, 448, 416, 402);
    k_gemm<<<gCV, 128, 0, stream>>>(CVh, 416, Wth, 416, net.bt, nullptr, 0,
                                    TLf, 416, nullptr, 0, 2000, 416, 402, 0);
    k_gemm<<<gCV, 128, 0, stream>>>(CVh, 416, Wgh, 416, net.bg, nullptr, 0,
                                    GLf, 416, nullptr, 0, 2000, 416, 402, 0);

    k_gate<<<blocks((size_t)2000 * 402, 256), 256, 0, stream>>>(TLf, GLf, GTf, 416, 2000, 402);
    k_final_gln<<<4, 256, 0, stream>>>(GTf, net.nw, net.nb, (float*)d_out, 416, 500, 402);
}
